// TLSTM_20143396618776
// MI455X (gfx1250) — 16-bit and 8-bit matrix operand order repaired and checked by reading, compiled, not run on hardware
//
#include <hip/hip_runtime.h>
#include <hip/hip_bf16.h>

#define B_ 256
#define S_ 2048
#define E_ 256
#define H_ 256
#define G_ 1024

typedef __attribute__((ext_vector_type(16))) __bf16 v16bf;
typedef __attribute__((ext_vector_type(8)))  float  v8f;
typedef __attribute__((ext_vector_type(8)))  int    v8i;

union V16 { v16bf bf; unsigned short us[16]; unsigned u[8]; uint4 q[2]; };
union V8  { v8i v; uint4 q[2]; unsigned u[8]; };

__device__ __forceinline__ unsigned short f2bf(float f) {
    union { __bf16 h; unsigned short u; } c;
    c.h = (__bf16)f;
    return c.u;
}
__device__ __forceinline__ unsigned pack2bf(float a, float b) {
    union { __bf16 h[2]; unsigned u; } c;
    c.h[0] = (__bf16)a;
    c.h[1] = (__bf16)b;
    return c.u;
}
__device__ __forceinline__ float bf2f(unsigned short b) {
    return __uint_as_float(((unsigned)b) << 16);
}
__device__ __forceinline__ float sigm(float x) {
    return 1.0f / (1.0f + __expf(-x));
}
__device__ __forceinline__ float tanh_fast(float x) {
    x = fminf(15.0f, fmaxf(-15.0f, x));
    float e = __expf(2.0f * x);
    return (e - 1.0f) / (e + 1.0f);
}
__device__ __forceinline__ unsigned f2e4m3_sw(float f) {
    unsigned u = __float_as_uint(f);
    unsigned s = (u >> 24) & 0x80u;
    unsigned a = u & 0x7fffffffu;
    if (a >= 0x43E00000u) return s | 0x7Eu;
    int e = (int)(a >> 23) - 127;
    if (e < -6) {
        float av = __uint_as_float(a);
        int v = (int)(av * 512.0f + 0.5f);
        if (v <= 0) return s;
        if (v >= 8) return s | 0x08u;
        return s | (unsigned)v;
    }
    unsigned m    = a & 0x7fffffu;
    unsigned m3   = m >> 20;
    unsigned rest = m & 0xFFFFFu;
    if (rest > 0x80000u || (rest == 0x80000u && (m3 & 1u))) {
        m3++;
        if (m3 == 8u) { m3 = 0u; e++; if (e > 8) return s | 0x7Eu; }
    }
    if (e == 8 && m3 == 7u) return s | 0x7Eu;
    return s | ((unsigned)(e + 7) << 3) | m3;
}
__device__ __forceinline__ unsigned pack4fp8(float a, float b, float c, float d) {
#if __has_builtin(__builtin_amdgcn_cvt_pk_fp8_f32)
    int v = 0;
    v = __builtin_amdgcn_cvt_pk_fp8_f32(a, b, v, false);
    v = __builtin_amdgcn_cvt_pk_fp8_f32(c, d, v, true);
    return (unsigned)v;
#else
    return f2e4m3_sw(a) | (f2e4m3_sw(b) << 8) | (f2e4m3_sw(c) << 16) |
           (f2e4m3_sw(d) << 24);
#endif
}

__device__ __forceinline__ v8f vzero8() {
    v8f z = {0.f, 0.f, 0.f, 0.f, 0.f, 0.f, 0.f, 0.f};
    return z;
}
__device__ __forceinline__ v8f wmma_fp8(v8i a, v8i b, v8f c) {
    return __builtin_amdgcn_wmma_f32_16x16x64_fp8_fp8(a, b, (short)0, c, false, false);
}

__global__ void prep_kernel(const float* __restrict__ Uw,
                            const float* __restrict__ Wa,
                            const float* __restrict__ Wd,
                            unsigned short* __restrict__ wU,
                            unsigned* __restrict__ wallF,
                            unsigned* __restrict__ wdF) {
    int idx = blockIdx.x * blockDim.x + threadIdx.x;
    if (idx < G_ * E_) { wU[idx] = f2bf(Uw[idx]); return; }
    idx -= G_ * E_;
    if (idx < 65536) {
        int f = idx >> 8, r = idx & 255, ln = r >> 3, v = r & 7;
        int nt = f >> 2, kt = f & 3;
        int n  = nt * 16 + (ln & 15);
        unsigned d = 0;
#pragma unroll
        for (int bb = 0; bb < 4; bb++) {
            int K = ((v < 4) ? 0 : 32) + ((ln >> 4) & 1) * 16 + (v & 3) * 4 + bb;
            d |= f2e4m3_sw(Wa[n * H_ + kt * 64 + K]) << (8 * bb);
        }
        wallF[idx] = d;
        return;
    }
    idx -= 65536;
    if (idx < 16384) {
        int f = idx >> 8, r = idx & 255, ln = r >> 3, v = r & 7;
        int nt = f >> 2, kt = f & 3;
        int n  = nt * 16 + (ln & 15);
        unsigned d = 0;
#pragma unroll
        for (int bb = 0; bb < 4; bb++) {
            int K = ((v < 4) ? 0 : 32) + ((ln >> 4) & 1) * 16 + (v & 3) * 4 + bb;
            d |= f2e4m3_sw(Wd[n * H_ + kt * 64 + K]) << (8 * bb);
        }
        wdF[idx] = d;
    }
}

__global__ void __launch_bounds__(256) proj_kernel(
        const float* __restrict__ x,
        const unsigned short* __restrict__ wU,
        const float* __restrict__ ub,
        unsigned short* __restrict__ uout) {
    const int lane = threadIdx.x & 31;
    const int wid  = blockIdx.x * (blockDim.x >> 5) + (threadIdx.x >> 5);
    const int mt   = wid >> 4;
    const int n64  = wid & 15;
    const int nlo  = lane & 15;
    const int hi   = (lane >> 4) & 1;

    const float* xrow = x + (size_t)(mt * 16 + nlo) * E_;

    v8f acc[4];
#pragma unroll
    for (int ft = 0; ft < 4; ft++) acc[ft] = vzero8();

#pragma unroll
    for (int kt = 0; kt < 8; kt++) {
        V16 a;
        const int kb = kt * 32 + hi * 16;
        const float4* p0 = (const float4*)(xrow + kb);
        const float4* p1 = (const float4*)(xrow + kb + 8);
        float4 f0 = p0[0], f1 = p0[1], f2 = p1[0], f3 = p1[1];
        a.u[0] = pack2bf(f0.x, f0.y);
        a.u[1] = pack2bf(f0.z, f0.w);
        a.u[2] = pack2bf(f1.x, f1.y);
        a.u[3] = pack2bf(f1.z, f1.w);
        a.u[4] = pack2bf(f2.x, f2.y);
        a.u[5] = pack2bf(f2.z, f2.w);
        a.u[6] = pack2bf(f3.x, f3.y);
        a.u[7] = pack2bf(f3.z, f3.w);
#pragma unroll
        for (int ft = 0; ft < 4; ft++) {
            const int g = n64 * 64 + ft * 16 + nlo;
            const v16bf* bp =
                (const v16bf*)(wU + (size_t)g * E_ + kt * 32 + hi * 16);
            acc[ft] = __builtin_amdgcn_wmma_f32_16x16x32_bf16(
                false, a.bf, false, *bp, (short)0, acc[ft], false, false);
        }
    }
#pragma unroll
    for (int ft = 0; ft < 4; ft++) {
        const int g    = n64 * 64 + ft * 16 + nlo;
        const int jcol = g & 255;
        const int gate = g >> 8;
        const float bi = ub[g];
#pragma unroll
        for (int r = 0; r < 8; r++) {
            const int row = mt * 16 + r + hi * 8;
            uout[(size_t)row * G_ + jcol * 4 + gate] = f2bf(acc[ft][r] + bi);
        }
    }
}

#define HSTRIDE 272
#define SW_BYTES 262144
#define SCAN_LDS (SW_BYTES + 2 * 16 * HSTRIDE + 64)

__global__ void __launch_bounds__(512) scan_kernel(
        const unsigned short* __restrict__ u_all,
        const float* __restrict__ ts,
        const unsigned* __restrict__ wallFrag,
        const unsigned* __restrict__ wdFrag,
        const float* __restrict__ wab,
        const float* __restrict__ wdbv,
        float* __restrict__ out) {
    extern __shared__ char smem[];
    unsigned* sW       = (unsigned*)smem;
    unsigned char* h8  = (unsigned char*)(smem + SW_BYTES);
    unsigned char* c8  = h8 + 16 * HSTRIDE;
    float* t_s         = (float*)(c8 + 16 * HSTRIDE);

    const int tid  = threadIdx.x;
    const int lane = tid & 31;
    const int w    = tid >> 5;
    const int b0   = blockIdx.x * 16;
    const int nlo  = lane & 15;
    const int hi   = (lane >> 4) & 1;

    {
        const uint4* src = (const uint4*)wallFrag;
        uint4* dst = (uint4*)sW;
        for (int i = tid; i < 65536 / 4; i += 512) dst[i] = src[i];
    }
    for (int i = tid; i < 16 * HSTRIDE; i += 512) { h8[i] = 0; c8[i] = 0; }

    V8 wd[4];
#pragma unroll
    for (int kt = 0; kt < 4; kt++) {
        const uint4* p = (const uint4*)(wdFrag + ((w * 4 + kt) * 256 + lane * 8));
        wd[kt].q[0] = p[0];
        wd[kt].q[1] = p[1];
    }
    float wb[4];
#pragma unroll
    for (int gb = 0; gb < 4; gb++) wb[gb] = wab[gb * 256 + w * 16 + nlo];
    const float wdb = wdbv[w * 16 + nlo];

    size_t urow[8], orow[8];
#pragma unroll
    for (int r = 0; r < 8; r++) {
        const int M = r + hi * 8;
        urow[r] = (size_t)(b0 + M) * S_ * (size_t)G_;
        orow[r] = (size_t)(b0 + M) * S_ * (size_t)H_;
    }
    const int jcol4 = (w * 16 + nlo) * 4;

    v8f c_f = vzero8();
    __syncthreads();

    for (int s = 0; s < S_; ++s) {
        if (tid < 16) t_s[tid] = ts[(size_t)(b0 + tid) * S_ + s];
        __syncthreads();

        v8f accD = vzero8();
#pragma unroll
        for (int kt = 0; kt < 4; kt++) {
            V8 a;
#pragma unroll
            for (int v = 0; v < 8; v++) {
                const int K = kt * 64 + (v >> 2) * 32 + hi * 16 + (v & 3) * 4;
                a.u[v] = *(const unsigned*)(c8 + nlo * HSTRIDE + K);
            }
            accD = wmma_fp8(a.v, wd[kt].v, accD);
        }

        v8f accG[4];
#pragma unroll
        for (int gb = 0; gb < 4; gb++) accG[gb] = vzero8();
#pragma unroll
        for (int kt = 0; kt < 4; kt++) {
            V8 a;
#pragma unroll
            for (int v = 0; v < 8; v++) {
                const int K = kt * 64 + (v >> 2) * 32 + hi * 16 + (v & 3) * 4;
                a.u[v] = *(const unsigned*)(h8 + nlo * HSTRIDE + K);
            }
#pragma unroll
            for (int gb = 0; gb < 4; gb++) {
                const int f = (gb * 16 + w) * 4 + kt;
                const uint4* bp = (const uint4*)(sW + f * 256 + lane * 8);
                V8 b;
                b.q[0] = bp[0];
                b.q[1] = bp[1];
                accG[gb] = wmma_fp8(a.v, b.v, accG[gb]);
            }
        }

        const size_t soff = (size_t)s * G_;
        float uv[4][8];
#pragma unroll
        for (int r = 0; r < 8; r++) {
            const uint2 uq =
                *(const uint2*)(u_all + urow[r] + soff + jcol4);
            uv[0][r] = bf2f((unsigned short)(uq.x & 0xffffu));
            uv[1][r] = bf2f((unsigned short)(uq.x >> 16));
            uv[2][r] = bf2f((unsigned short)(uq.y & 0xffffu));
            uv[3][r] = bf2f((unsigned short)(uq.y >> 16));
        }
        if (s + 1 < S_) {
#pragma unroll
            for (int r = 0; r < 8; r += 2)
                __builtin_prefetch(u_all + urow[r] + soff + G_ + jcol4, 0, 1);
        }

        float hn[8], cn[8];
#pragma unroll
        for (int r = 0; r < 8; r++) {
            const int M   = r + hi * 8;
            const float cs1  = tanh_fast(accD[r] + wdb);
            const float cadj = c_f[r] + cs1 * (t_s[M] - 1.0f);
            const float fg = sigm(accG[0][r] + wb[0] + uv[0][r]);
            const float ig = sigm(accG[1][r] + wb[1] + uv[1][r]);
            const float og = sigm(accG[2][r] + wb[2] + uv[2][r]);
            const float cg = sigm(accG[3][r] + wb[3] + uv[3][r]);
            const float c2 = fg * cadj + ig * cg;
            const float h2 = og * tanh_fast(c2);
            c_f[r] = c2;
            cn[r]  = c2;
            hn[r]  = h2;
            out[orow[r] + (size_t)s * H_ + w * 16 + nlo] = h2;
        }

        __syncthreads();
        const unsigned hp0 = pack4fp8(hn[0], hn[1], hn[2], hn[3]);
        const unsigned hp1 = pack4fp8(hn[4], hn[5], hn[6], hn[7]);
        const unsigned cp0 = pack4fp8(cn[0], cn[1], cn[2], cn[3]);
        const unsigned cp1 = pack4fp8(cn[4], cn[5], cn[6], cn[7]);
        const int colb = w * 16 + nlo;
#pragma unroll
        for (int r = 0; r < 8; r++) {
            const int M = r + hi * 8;
            const unsigned hv = (r < 4) ? hp0 : hp1;
            const unsigned cv = (r < 4) ? cp0 : cp1;
            h8[M * HSTRIDE + colb] = (unsigned char)((hv >> (8 * (r & 3))) & 0xffu);
            c8[M * HSTRIDE + colb] = (unsigned char)((cv >> (8 * (r & 3))) & 0xffu);
        }
    }
}

extern "C" void kernel_launch(void* const* d_in, const int* in_sizes, int n_in,
                              void* d_out, int out_size, void* d_ws, size_t ws_size,
                              hipStream_t stream) {
    const float* inputs     = (const float*)d_in[0];
    const float* timestamps = (const float*)d_in[1];
    const float* W_all_w    = (const float*)d_in[2];
    const float* W_all_b    = (const float*)d_in[3];
    const float* U_all_w    = (const float*)d_in[4];
    const float* U_all_b    = (const float*)d_in[5];
    const float* W_d_w      = (const float*)d_in[6];
    const float* W_d_b      = (const float*)d_in[7];
    float* out              = (float*)d_out;

    char* ws = (char*)d_ws;
    const size_t UALL_BYTES = (size_t)B_ * S_ * G_ * 2;
    unsigned short* u_all = (unsigned short*)ws;
    unsigned short* wU    = (unsigned short*)(ws + UALL_BYTES);
    unsigned*       wallF = (unsigned*)(ws + UALL_BYTES + (size_t)G_ * E_ * 2);
    unsigned*       wdF   = wallF + 65536;

    {
        const int total = G_ * E_ + 65536 + 16384;
        prep_kernel<<<(total + 255) / 256, 256, 0, stream>>>(
            U_all_w, W_all_w, W_d_w, wU, wallF, wdF);
    }
    {
        const int mtiles = (B_ * S_) / 16;
        const int waves  = mtiles * 16;
        proj_kernel<<<waves / 8, 256, 0, stream>>>(inputs, wU, U_all_b, u_all);
    }
    scan_kernel<<<B_ / 16, 512, SCAN_LDS, stream>>>(
        u_all, timestamps, wallF, wdF, W_all_b, W_d_b, out);
}
